// GroupedQueryAttention_48189533061443
// MI455X (gfx1250) — matrix operand order repaired and checked by reading, not compiled, not run on hardware
//
#include <hip/hip_runtime.h>
#include <hip/hip_bf16.h>
#include <math.h>

// ---------------------------------------------------------------------------
// GroupedQueryAttention for MI455X (gfx1250, wave32, WMMA 16x16x32 bf16)
//   B=2, T=2048, HIDDEN=2048, NUM_Q=32, NUM_KV=8, HEAD_DIM=64
// ---------------------------------------------------------------------------

typedef __attribute__((ext_vector_type(16))) __bf16 v16bf;
typedef __attribute__((ext_vector_type(8)))  float  v8f;

struct alignas(8) bf16x4 { __bf16 x, y, z, w; };

#define TILE_M 128
#define TILE_N 128
#define TILE_K 32
#define LDA    (TILE_K + 8)             // bf16 elems per LDS row (80B rows)
#define LDS_A_BYTES  (TILE_M * LDA * 2) // 10240
#define GEMM_SMEM    (TILE_M * TILE_N * 4) // 65536: C tile aliases A/BT tiles

// A-fragment K-base for 16-bit A matrix (contiguous k order, matches B):
// VGPR v holds K pair {base, base+1}; lanes 16-31 add 16 to K.
__device__ __forceinline__ int a_kbase(int v, int half) {
    return 2 * v + 16 * half;
}

// ---------------------------------------------------------------------------
// GEMM: C[M,N] = A[M,K] * B[K,N], fp32 in/out, bf16 WMMA compute.
// 256 thr = 8 waves -> 128x128 C tile; wave = 32x64 (2x4 WMMA subtiles).
// Scalar base + 32-bit offsets (saddr form), clause-friendly split staging,
// LDS-staged float4 writeback, next-tile prefetch.
// ---------------------------------------------------------------------------
__global__ __launch_bounds__(256)
void gemm_bf16_wmma(const float* __restrict__ A, const float* __restrict__ B,
                    float* __restrict__ C, int M, int N, int K)
{
    extern __shared__ char smem[];
    __bf16 (*lds_a )[LDA]    = (__bf16(*)[LDA])(smem);                // [m][k]
    __bf16 (*lds_bt)[LDA]    = (__bf16(*)[LDA])(smem + LDS_A_BYTES);  // [n][k]
    float  (*lds_c )[TILE_N] = (float (*)[TILE_N])(smem);             // aliases

    const int tid  = threadIdx.x;
    const int lane = tid & 31;
    const int wave = tid >> 5;
    const int wm   = wave & 3;    // 4 waves along M
    const int wn   = wave >> 2;   // 2 waves along N
    const int half = lane >> 4;
    const int l15  = lane & 15;

    const int m0 = blockIdx.x * TILE_M;
    const int n0 = blockIdx.y * TILE_N;

    // Uniform bases (SGPRs) + per-thread 32-bit offsets (precomputed once).
    const float* aBase = A + (size_t)m0 * K;        // + k0 per iter (scalar)
    const float* bBase = B + n0;                    // + k0*N per iter (scalar)
    int aRow[4], aOff[4], bRow[4], bCol[4], bOff[4];
    #pragma unroll
    for (int i = 0; i < 4; ++i) {
        int fa = tid + i * 256;
        aRow[i] = fa >> 3;  aOff[i] = aRow[i] * K + (fa & 7) * 4;
        int fb = tid + i * 256;
        bRow[i] = fb >> 5;  bCol[i] = (fb & 31) * 4;
        bOff[i] = bRow[i] * N + bCol[i];
    }

    v8f acc[2][4];
    #pragma unroll
    for (int mi = 0; mi < 2; ++mi)
        #pragma unroll
        for (int ni = 0; ni < 4; ++ni)
            acc[mi][ni] = (v8f){0.f,0.f,0.f,0.f,0.f,0.f,0.f,0.f};

    for (int k0 = 0; k0 < K; k0 += TILE_K) {
        const float* aP = aBase + k0;               // uniform
        const float* bP = bBase + (size_t)k0 * N;   // uniform

        // ---- load phase: 8 independent b128 loads, one clause, one wait
        float4 aReg[4], bReg[4];
        #pragma unroll
        for (int i = 0; i < 4; ++i) aReg[i] = *(const float4*)(aP + aOff[i]);
        #pragma unroll
        for (int i = 0; i < 4; ++i) bReg[i] = *(const float4*)(bP + bOff[i]);

        // Prefetch next k-chunk (global_prefetch_b8).
        if (k0 + TILE_K < K) {
            __builtin_prefetch(aP + TILE_K + aOff[0], 0, 1);
            __builtin_prefetch(bP + (size_t)TILE_K * N + bOff[0], 0, 1);
        }

        // ---- convert/store phase
        #pragma unroll
        for (int i = 0; i < 4; ++i) {
            int fa = tid + i * 256;
            bf16x4 w = {(__bf16)aReg[i].x, (__bf16)aReg[i].y,
                        (__bf16)aReg[i].z, (__bf16)aReg[i].w};
            *(bf16x4*)&lds_a[aRow[i]][(fa & 7) * 4] = w;
        }
        #pragma unroll
        for (int i = 0; i < 4; ++i) {
            lds_bt[bCol[i] + 0][bRow[i]] = (__bf16)bReg[i].x;
            lds_bt[bCol[i] + 1][bRow[i]] = (__bf16)bReg[i].y;
            lds_bt[bCol[i] + 2][bRow[i]] = (__bf16)bReg[i].z;
            lds_bt[bCol[i] + 3][bRow[i]] = (__bf16)bReg[i].w;
        }
        __syncthreads();

        v16bf afrag[2], bfrag[4];
        #pragma unroll
        for (int mi = 0; mi < 2; ++mi) {
            const __bf16* arow = &lds_a[wm * 32 + mi * 16 + l15][0];
            #pragma unroll
            for (int v = 0; v < 8; ++v) {
                int kb = a_kbase(v, half);
                afrag[mi][2 * v]     = arow[kb];
                afrag[mi][2 * v + 1] = arow[kb + 1];
            }
        }
        #pragma unroll
        for (int ni = 0; ni < 4; ++ni) {
            const __bf16* bcol = &lds_bt[wn * 64 + ni * 16 + l15][16 * half];
            #pragma unroll
            for (int i = 0; i < 16; ++i) bfrag[ni][i] = bcol[i];
        }

        #pragma unroll
        for (int mi = 0; mi < 2; ++mi)
            #pragma unroll
            for (int ni = 0; ni < 4; ++ni)
                acc[mi][ni] = __builtin_amdgcn_wmma_f32_16x16x32_bf16(
                    false, afrag[mi], false, bfrag[ni],
                    (short)0, acc[mi][ni], false, false);

        __syncthreads();
    }

    // Writeback: stage C tile in LDS (aliases A/BT), then coalesced float4
    // with uniform base + 32-bit offsets.
    #pragma unroll
    for (int mi = 0; mi < 2; ++mi)
        #pragma unroll
        for (int ni = 0; ni < 4; ++ni)
            #pragma unroll
            for (int v = 0; v < 8; ++v)
                lds_c[wm * 32 + mi * 16 + v + 8 * half]
                     [wn * 64 + ni * 16 + l15] = acc[mi][ni][v];
    __syncthreads();
    float* cBase = C + (size_t)m0 * N + n0;         // uniform
    #pragma unroll
    for (int i = 0; i < 16; ++i) {
        int flat4 = tid + i * 256;                  // 4096 float4 total
        int r = flat4 >> 5, c4 = (flat4 & 31) * 4;
        *(float4*)(cBase + r * N + c4) = *(const float4*)&lds_c[r][c4];
    }
}

// ---------------------------------------------------------------------------
// Flash-style GQA attention. 128 thr = 4 waves per (b, h, 64-query block);
// waves share LDS-staged K/V chunks (K as [key][d], V transposed [d][key]
// so both B-fragments are contiguous -> ds_load_b128).
//   Q:[B,T,32,64]  K,V:[B,T,8,64]  O:[B,T,32,64]   (all fp32)
// ---------------------------------------------------------------------------
#define ATT_T   2048
#define ATT_HQ  32
#define ATT_HKV 8

__global__ __launch_bounds__(128)
void gqa_flash_wmma(const float* __restrict__ Q, const float* __restrict__ K,
                    const float* __restrict__ V, float* __restrict__ O)
{
    __shared__ __bf16 lds_k [32][72];      // [key][d], 64 + 8 pad
    __shared__ __bf16 lds_vt[64][40];      // [d][key], 32 + 8 pad
    __shared__ __bf16 p_sm[4][16][40];     // per-wave P tile (C->A transpose)

    const int tid  = threadIdx.x;
    const int lane = tid & 31;
    const int wave = tid >> 5;
    const int half = lane >> 4;
    const int l15  = lane & 15;

    const int nqb = ATT_T / 64;                    // 32 query blocks per head
    const int qb = blockIdx.x % nqb;
    const int h  = (blockIdx.x / nqb) % ATT_HQ;
    const int b  =  blockIdx.x / (nqb * ATT_HQ);
    const int kv = h >> 2;                         // group = 32/8 = 4
    const int q0 = qb * 64 + wave * 16;            // this wave's query tile

    const float scale = 0.125f;                    // 1/sqrt(64)

    // Per-thread 32-bit K/V offsets within a 32-key chunk.
    int kvOff[4], kKey[4], kD4[4];
    #pragma unroll
    for (int i = 0; i < 4; ++i) {
        int flat4 = tid + i * 128;
        kKey[i] = flat4 >> 4;
        kD4[i]  = (flat4 & 15) * 4;
        kvOff[i] = kKey[i] * (ATT_HKV * 64) + kD4[i];
    }
    // Uniform K/V chunk bases (advanced by scalar stride per chunk).
    const size_t chunkStride = (size_t)32 * ATT_HKV * 64;
    const float* kBase = K + ((size_t)(b * ATT_T) * ATT_HKV + kv) * 64;
    const float* vBase = V + ((size_t)(b * ATT_T) * ATT_HKV + kv) * 64;

    // Q fragment (A layout), pre-scaled; query row m = l15.
    const size_t qrow = ((size_t)(b * ATT_T + q0 + l15) * ATT_HQ + h) * 64;
    v16bf aq[2];
    #pragma unroll
    for (int p = 0; p < 2; ++p)
        #pragma unroll
        for (int v = 0; v < 8; ++v) {
            int kb = a_kbase(v, half);
            aq[p][2 * v]     = (__bf16)(Q[qrow + p * 32 + kb]     * scale);
            aq[p][2 * v + 1] = (__bf16)(Q[qrow + p * 32 + kb + 1] * scale);
        }

    float run_max[8], run_sum[8];
    v8f o_acc[4];
    #pragma unroll
    for (int v = 0; v < 8; ++v) { run_max[v] = -3.0e38f; run_sum[v] = 0.f; }
    #pragma unroll
    for (int ni = 0; ni < 4; ++ni)
        o_acc[ni] = (v8f){0.f,0.f,0.f,0.f,0.f,0.f,0.f,0.f};

    for (int jc = 0; jc < ATT_T / 32; ++jc) {
        const float* kP = kBase + jc * chunkStride;   // uniform
        const float* vP = vBase + jc * chunkStride;   // uniform

        __syncthreads();   // previous chunk fully consumed
        // ---- load phase: 8 independent b128 loads
        float4 kReg[4], vReg[4];
        #pragma unroll
        for (int i = 0; i < 4; ++i) kReg[i] = *(const float4*)(kP + kvOff[i]);
        #pragma unroll
        for (int i = 0; i < 4; ++i) vReg[i] = *(const float4*)(vP + kvOff[i]);
        // ---- convert/store phase
        #pragma unroll
        for (int i = 0; i < 4; ++i) {
            bf16x4 kw = {(__bf16)kReg[i].x, (__bf16)kReg[i].y,
                         (__bf16)kReg[i].z, (__bf16)kReg[i].w};
            *(bf16x4*)&lds_k[kKey[i]][kD4[i]] = kw;
            lds_vt[kD4[i] + 0][kKey[i]] = (__bf16)vReg[i].x;
            lds_vt[kD4[i] + 1][kKey[i]] = (__bf16)vReg[i].y;
            lds_vt[kD4[i] + 2][kKey[i]] = (__bf16)vReg[i].z;
            lds_vt[kD4[i] + 3][kKey[i]] = (__bf16)vReg[i].w;
        }
        __syncthreads();

        // ---- S = Q * K^T : two 16x16 key tiles, d = 64 (2 WMMA each)
        v8f s[2];
        #pragma unroll
        for (int t = 0; t < 2; ++t) {
            s[t] = (v8f){0.f,0.f,0.f,0.f,0.f,0.f,0.f,0.f};
            #pragma unroll
            for (int p = 0; p < 2; ++p) {
                const __bf16* kp = &lds_k[t * 16 + l15][p * 32 + 16 * half];
                v16bf bk;
                #pragma unroll
                for (int i = 0; i < 16; ++i) bk[i] = kp[i];
                s[t] = __builtin_amdgcn_wmma_f32_16x16x32_bf16(
                    false, aq[p], false, bk, (short)0, s[t], false, false);
            }
        }

        // ---- online softmax (row m = v + 8*half, spread over 16 lanes)
        #pragma unroll
        for (int v = 0; v < 8; ++v) {
            float mx = fmaxf(s[0][v], s[1][v]);
            #pragma unroll
            for (int m = 1; m <= 8; m <<= 1)
                mx = fmaxf(mx, __shfl_xor(mx, m, 32));
            float nmax = fmaxf(run_max[v], mx);
            float corr = __expf(run_max[v] - nmax);
            run_max[v] = nmax;

            float e0 = __expf(s[0][v] - nmax);
            float e1 = __expf(s[1][v] - nmax);
            float rs = e0 + e1;
            #pragma unroll
            for (int m = 1; m <= 8; m <<= 1)
                rs += __shfl_xor(rs, m, 32);
            run_sum[v] = run_sum[v] * corr + rs;

            #pragma unroll
            for (int ni = 0; ni < 4; ++ni)
                o_acc[ni][v] *= corr;

            p_sm[wave][v + 8 * half][l15]      = (__bf16)e0;   // C->A transpose
            p_sm[wave][v + 8 * half][16 + l15] = (__bf16)e1;
        }
        // per-wave LDS RAW: ordered by in-order DS + compiler waits

        // ---- load P as A fragment (16 q x 32 keys)
        v16bf ap;
        #pragma unroll
        for (int v = 0; v < 8; ++v) {
            int kb = a_kbase(v, half);
            ap[2 * v]     = p_sm[wave][l15][kb];
            ap[2 * v + 1] = p_sm[wave][l15][kb + 1];
        }

        // ---- O += P * V : 4 d-subtiles of 16, K-dim = 32 keys
        #pragma unroll
        for (int ni = 0; ni < 4; ++ni) {
            const __bf16* vp = &lds_vt[ni * 16 + l15][16 * half];
            v16bf bv;
            #pragma unroll
            for (int i = 0; i < 16; ++i) bv[i] = vp[i];
            o_acc[ni] = __builtin_amdgcn_wmma_f32_16x16x32_bf16(
                false, ap, false, bv, (short)0, o_acc[ni], false, false);
        }
    }

    // ---- normalize and write O[b, q0+m, h, d]: uniform base + int offsets
    float* oBase = O + ((size_t)(b * ATT_T + q0) * ATT_HQ + h) * 64;
    #pragma unroll
    for (int v = 0; v < 8; ++v) {
        float rcp = 1.0f / run_sum[v];
        int orow = (v + 8 * half) * (ATT_HQ * 64);
        #pragma unroll
        for (int ni = 0; ni < 4; ++ni)
            oBase[orow + ni * 16 + l15] = o_acc[ni][v] * rcp;
    }
}

// ---------------------------------------------------------------------------
// Launch: Q/K/V projections -> attention -> output projection.
// ---------------------------------------------------------------------------
extern "C" void kernel_launch(void* const* d_in, const int* in_sizes, int n_in,
                              void* d_out, int out_size, void* d_ws, size_t ws_size,
                              hipStream_t stream)
{
    const float* x  = (const float*)d_in[0];
    const float* wq = (const float*)d_in[1];
    const float* wk = (const float*)d_in[2];
    const float* wv = (const float*)d_in[3];
    const float* wo = (const float*)d_in[4];
    float* out = (float*)d_out;

    const int M = 2 * 2048;      // B*T
    const int H = 2048;          // HIDDEN
    const int NQ = 32 * 64;      // 2048
    const int NKV = 8 * 64;      // 512

    float* qb = (float*)d_ws;                       //  4096*2048
    float* kb = qb + (size_t)M * NQ;                //  4096*512
    float* vb = kb + (size_t)M * NKV;               //  4096*512
    float* ab = vb + (size_t)M * NKV;               //  4096*2048

    dim3 blk(256);
    gemm_bf16_wmma<<<dim3(M / TILE_M, NQ  / TILE_N), blk, GEMM_SMEM, stream>>>(x, wq, qb, M, NQ,  H);
    gemm_bf16_wmma<<<dim3(M / TILE_M, NKV / TILE_N), blk, GEMM_SMEM, stream>>>(x, wk, kb, M, NKV, H);
    gemm_bf16_wmma<<<dim3(M / TILE_M, NKV / TILE_N), blk, GEMM_SMEM, stream>>>(x, wv, vb, M, NKV, H);

    gqa_flash_wmma<<<dim3(2 * 32 * (2048 / 64)), dim3(128), 0, stream>>>(qb, kb, vb, ab);

    gemm_bf16_wmma<<<dim3(M / TILE_M, H / TILE_N), blk, GEMM_SMEM, stream>>>(ab, wo, out, M, H, H);
}
